// NodeModel_17497696764457
// MI455X (gfx1250) — matrix operand order repaired and checked by reading, not compiled, not run on hardware
//
#include <hip/hip_runtime.h>
#include <hip/hip_bf16.h>

typedef __attribute__((ext_vector_type(16))) _Float16 v16h;
typedef __attribute__((ext_vector_type(8)))  _Float16 v8h;
typedef __attribute__((ext_vector_type(8)))  float    v8f;

#define NN   50000
#define NE   800000
#define FN   128
#define FE   64
#define FG   32
#define HID  256
#define TGT  128
#define IN1  192   // FN + FE
#define IN2  416   // FN + HID + FG

// ---------------------------------------------------------------------------
// Fragment helpers (CDNA5 WMMA 16x16x32 f16 layouts, wave32)
// A (16x32, MxK): lane L holds row M=L%16; K halves are
//   [kbase8 .. kbase8+7] and [16+kbase8 .. 16+kbase8+7], kbase8 = 8*(L/16)
// B (32x16, KxN): lane L holds col N=L%16; K = (L/16)*8 + (j%8) + (j/8)*16, j=0..15
// C (16x16 f32): lane L holds N=L%16; VGPR i holds M = i + 8*(L/16)
// ---------------------------------------------------------------------------
__device__ __forceinline__ v16h cat16(v8h lo, v8h hi) {
  return __builtin_shufflevector(lo, hi, 0, 1, 2, 3, 4, 5, 6, 7,
                                 8, 9, 10, 11, 12, 13, 14, 15);
}

__device__ __forceinline__ v16h load_a_frag(const _Float16* base, int strideH,
                                            int row, int kbase, int half) {
  const _Float16* p = base + row * strideH + kbase + half * 8;
  v8h lo = *(const v8h*)p;          // K = kbase8 .. kbase8+7
  v8h hi = *(const v8h*)(p + 16);   // K = 16+kbase8 .. 16+kbase8+7
  return cat16(lo, hi);
}

#define WMMA_F16(a, b, c) \
  __builtin_amdgcn_wmma_f32_16x16x32_f16(false, (a), false, (b), (short)0, (c), false, false)

// ---------------------------------------------------------------------------
// Workspace layout (bytes)
// ---------------------------------------------------------------------------
static constexpr size_t SUMMED_BYTES = (size_t)NN * HID * 4;   // 51,200,000
static constexpr size_t COUNT_BYTES  = (size_t)NN * 4;         //    200,000
static constexpr size_t OFF_COUNT = SUMMED_BYTES;              // contiguous after summed
static constexpr size_t OFF_W1 = ((SUMMED_BYTES + COUNT_BYTES + 255) / 256) * 256;
static constexpr size_t W1_HALVES = (size_t)(IN1 / 32) * (HID / 16) * 32 * 16;  //  49,152
static constexpr size_t OFF_W2 = OFF_W1 + W1_HALVES * 2;
static constexpr size_t W2_HALVES = (size_t)(HID / 32) * (HID / 16) * 32 * 16;  //  65,536
static constexpr size_t OFF_W3 = OFF_W2 + W2_HALVES * 2;
static constexpr size_t W3_HALVES = (size_t)(IN2 / 32) * (HID / 16) * 32 * 16;  // 106,496
static constexpr size_t OFF_W4 = OFF_W3 + W3_HALVES * 2;
static constexpr size_t W4_HALVES = (size_t)(HID / 32) * (TGT / 16) * 32 * 16;  //  32,768

// ---------------------------------------------------------------------------
// Utility kernels
// ---------------------------------------------------------------------------
__global__ void zero_f32_kernel(float* __restrict__ p, long n) {
  long i = (long)blockIdx.x * blockDim.x + threadIdx.x;
  if (i < n) p[i] = 0.0f;
}

// Pack row-major f32 weight [K][N] into WMMA B-fragment order:
// out[((kt*nTiles + nt)*32 + lane)*16 + j] = W[kt*32 + (lane/16)*8 + (j%8) + (j/8)*16][nt*16 + lane%16]
__global__ void pack_weight_kernel(const float* __restrict__ W, _Float16* __restrict__ out,
                                   int N, int nTiles, int total) {
  int idx = blockIdx.x * 256 + threadIdx.x;
  if (idx >= total) return;
  int j    = idx & 15;
  int lane = (idx >> 4) & 31;
  int t    = idx >> 9;
  int nt   = t % nTiles;
  int kt   = t / nTiles;
  int k = kt * 32 + (lane >> 4) * 8 + (j & 7) + (j >> 3) * 16;
  int n = nt * 16 + (lane & 15);
  out[idx] = (_Float16)W[(long)k * N + n];
}

// ---------------------------------------------------------------------------
// Edge MLP + scatter-add.  64 edges per block (4 M-subtiles of 16), 8 waves;
// wave w owns output columns [w*32, w*32+32) as two 16-wide N tiles.  Each B
// fragment load is reused across 4 WMMAs (4 M-subtiles).
// ---------------------------------------------------------------------------
#define EM      64          // edges per block
#define EMT     (EM / 16)   // 4 M-subtiles
#define EA_STR  (IN1 + 8)   // 200 halves, 400B rows (16B aligned)
#define H_STR   (HID + 8)   // 264 halves, 528B rows

__global__ __launch_bounds__(256) void edge_mlp_scatter_kernel(
    const float* __restrict__ x, const int* __restrict__ ei,
    const float* __restrict__ ea,
    const _Float16* __restrict__ W1p, const float* __restrict__ b1,
    const _Float16* __restrict__ W2p, const float* __restrict__ b2,
    float* __restrict__ summed, float* __restrict__ count) {
  __shared__ _Float16 Ash[EM * EA_STR];   // 25,600 B
  __shared__ _Float16 Hsh[EM * H_STR];    // 33,792 B
  __shared__ int rows_s[EM];
  __shared__ int cols_s[EM];

  const int tid  = threadIdx.x;
  const int wave = tid >> 5;
  const int lane = tid & 31;
  const int half = lane >> 4;
  const int ln16 = lane & 15;
  const long e0  = (long)blockIdx.x * EM;

  if (tid < EM) {
    rows_s[tid] = ei[e0 + tid];
    cols_s[tid] = ei[(long)NE + e0 + tid];
  }
  __syncthreads();

  // Build A tile: [64 edges][x[row] (128) | edge_attr (64)] as f16
  for (int idx = tid; idx < EM * IN1; idx += 256) {
    int r = idx / IN1, c = idx - r * IN1;
    float v = (c < FN) ? x[(long)rows_s[r] * FN + c]
                       : ea[(e0 + r) * FE + (c - FN)];
    Ash[r * EA_STR + c] = (_Float16)v;
  }
  if (tid < EM) atomicAdd(&count[cols_s[tid]], 1.0f);
  __syncthreads();

  const int nt0 = wave * 2, nt1 = wave * 2 + 1;

  // ----- Layer 1: [64 x 192] @ [192 x 256] + b1, ReLU -----
  v8f acc[EMT][2] = {};
  for (int kt = 0; kt < IN1 / 32; ++kt) {
    v16h bf0 = *(const v16h*)(W1p + ((((long)kt * 16 + nt0) * 32 + lane) << 4));
    v16h bf1 = *(const v16h*)(W1p + ((((long)kt * 16 + nt1) * 32 + lane) << 4));
#pragma unroll
    for (int mt = 0; mt < EMT; ++mt) {
      v16h a = load_a_frag(Ash, EA_STR, mt * 16 + ln16, kt * 32, half);
      acc[mt][0] = WMMA_F16(a, bf0, acc[mt][0]);
      acc[mt][1] = WMMA_F16(a, bf1, acc[mt][1]);
    }
  }
  {
    float biasA = b1[nt0 * 16 + ln16];
    float biasB = b1[nt1 * 16 + ln16];
#pragma unroll
    for (int mt = 0; mt < EMT; ++mt)
#pragma unroll
      for (int i = 0; i < 8; ++i) {
        int m = mt * 16 + i + 8 * half;
        Hsh[m * H_STR + nt0 * 16 + ln16] = (_Float16)fmaxf(acc[mt][0][i] + biasA, 0.0f);
        Hsh[m * H_STR + nt1 * 16 + ln16] = (_Float16)fmaxf(acc[mt][1][i] + biasB, 0.0f);
      }
  }
  __syncthreads();

  // ----- Layer 2: [64 x 256] @ [256 x 256] + b2, scatter-add to summed -----
  v8f acc2[EMT][2] = {};
  for (int kt = 0; kt < HID / 32; ++kt) {
    v16h bf0 = *(const v16h*)(W2p + ((((long)kt * 16 + nt0) * 32 + lane) << 4));
    v16h bf1 = *(const v16h*)(W2p + ((((long)kt * 16 + nt1) * 32 + lane) << 4));
#pragma unroll
    for (int mt = 0; mt < EMT; ++mt) {
      v16h a = load_a_frag(Hsh, H_STR, mt * 16 + ln16, kt * 32, half);
      acc2[mt][0] = WMMA_F16(a, bf0, acc2[mt][0]);
      acc2[mt][1] = WMMA_F16(a, bf1, acc2[mt][1]);
    }
  }
  {
    float c0 = b2[nt0 * 16 + ln16];
    float c1 = b2[nt1 * 16 + ln16];
#pragma unroll
    for (int mt = 0; mt < EMT; ++mt)
#pragma unroll
      for (int i = 0; i < 8; ++i) {
        int m = mt * 16 + i + 8 * half;
        long dst = cols_s[m];
        atomicAdd(&summed[dst * HID + nt0 * 16 + ln16], acc2[mt][0][i] + c0);
        atomicAdd(&summed[dst * HID + nt1 * 16 + ln16], acc2[mt][1][i] + c1);
      }
  }
}

// ---------------------------------------------------------------------------
// Node MLP: z = [x | summed/count | u[batch]] -> relu(z@W3+b3) @ W4 + b4
// 32 nodes per block (Z tile is 416 wide; 64 rows would exceed static LDS).
// ---------------------------------------------------------------------------
#define Z_STR (IN2 + 8)   // 424 halves, 848B rows (16B aligned)

__global__ __launch_bounds__(256) void node_mlp_kernel(
    const float* __restrict__ x, const float* __restrict__ u,
    const int* __restrict__ batch,
    const float* __restrict__ summed, const float* __restrict__ count,
    const _Float16* __restrict__ W3p, const float* __restrict__ b3,
    const _Float16* __restrict__ W4p, const float* __restrict__ b4,
    float* __restrict__ out) {
  __shared__ _Float16 Zsh[32 * Z_STR];
  __shared__ _Float16 Hsh[32 * H_STR];

  const int tid  = threadIdx.x;
  const int wave = tid >> 5;
  const int lane = tid & 31;
  const int half = lane >> 4;
  const int ln16 = lane & 15;
  const long n0  = (long)blockIdx.x * 32;

  // Build Z tile (clamp out-of-range rows; their stores are suppressed later)
  for (int idx = tid; idx < 32 * IN2; idx += 256) {
    int r = idx / IN2, c = idx - r * IN2;
    long node = n0 + r;
    if (node >= NN) node = NN - 1;
    float v;
    if (c < FN) {
      v = x[node * FN + c];
    } else if (c < FN + HID) {
      float cnt = count[node];
      v = summed[node * HID + (c - FN)] / fmaxf(cnt, 1.0f);
    } else {
      v = u[(long)batch[node] * FG + (c - FN - HID)];
    }
    Zsh[r * Z_STR + c] = (_Float16)v;
  }
  __syncthreads();

  const int nt0 = wave * 2, nt1 = wave * 2 + 1;

  // ----- Layer 3: [32 x 416] @ [416 x 256] + b3, ReLU -----
  v8f acc[2][2] = {};
  for (int kt = 0; kt < IN2 / 32; ++kt) {
    v16h bf0 = *(const v16h*)(W3p + ((((long)kt * 16 + nt0) * 32 + lane) << 4));
    v16h bf1 = *(const v16h*)(W3p + ((((long)kt * 16 + nt1) * 32 + lane) << 4));
#pragma unroll
    for (int mt = 0; mt < 2; ++mt) {
      v16h a = load_a_frag(Zsh, Z_STR, mt * 16 + ln16, kt * 32, half);
      acc[mt][0] = WMMA_F16(a, bf0, acc[mt][0]);
      acc[mt][1] = WMMA_F16(a, bf1, acc[mt][1]);
    }
  }
  {
    float biasA = b3[nt0 * 16 + ln16];
    float biasB = b3[nt1 * 16 + ln16];
#pragma unroll
    for (int mt = 0; mt < 2; ++mt)
#pragma unroll
      for (int i = 0; i < 8; ++i) {
        int m = mt * 16 + i + 8 * half;
        Hsh[m * H_STR + nt0 * 16 + ln16] = (_Float16)fmaxf(acc[mt][0][i] + biasA, 0.0f);
        Hsh[m * H_STR + nt1 * 16 + ln16] = (_Float16)fmaxf(acc[mt][1][i] + biasB, 0.0f);
      }
  }
  __syncthreads();

  // ----- Layer 4: [32 x 256] @ [256 x 128] + b4 -> out -----
  v8f acc4[2] = {};
  for (int kt = 0; kt < HID / 32; ++kt) {
    v16h bf = *(const v16h*)(W4p + ((((long)kt * 8 + wave) * 32 + lane) << 4));
#pragma unroll
    for (int mt = 0; mt < 2; ++mt) {
      v16h a = load_a_frag(Hsh, H_STR, mt * 16 + ln16, kt * 32, half);
      acc4[mt] = WMMA_F16(a, bf, acc4[mt]);
    }
  }
  {
    float bb = b4[wave * 16 + ln16];
#pragma unroll
    for (int mt = 0; mt < 2; ++mt)
#pragma unroll
      for (int i = 0; i < 8; ++i) {
        long node = n0 + mt * 16 + i + 8 * half;
        if (node < NN) out[node * TGT + wave * 16 + ln16] = acc4[mt][i] + bb;
      }
  }
}

// ---------------------------------------------------------------------------
// Host launch
// ---------------------------------------------------------------------------
extern "C" void kernel_launch(void* const* d_in, const int* in_sizes, int n_in,
                              void* d_out, int out_size, void* d_ws, size_t ws_size,
                              hipStream_t stream) {
  const float* x     = (const float*)d_in[0];
  const int*   ei    = (const int*)d_in[1];
  const float* ea    = (const float*)d_in[2];
  const float* u     = (const float*)d_in[3];
  const int*   batch = (const int*)d_in[4];
  const float* W1 = (const float*)d_in[5];
  const float* b1 = (const float*)d_in[6];
  const float* W2 = (const float*)d_in[7];
  const float* b2 = (const float*)d_in[8];
  const float* W3 = (const float*)d_in[9];
  const float* b3 = (const float*)d_in[10];
  const float* W4 = (const float*)d_in[11];
  const float* b4 = (const float*)d_in[12];
  float* out = (float*)d_out;

  char* ws = (char*)d_ws;
  float*    summed = (float*)ws;
  float*    count  = (float*)(ws + OFF_COUNT);
  _Float16* W1p    = (_Float16*)(ws + OFF_W1);
  _Float16* W2p    = (_Float16*)(ws + OFF_W2);
  _Float16* W3p    = (_Float16*)(ws + OFF_W3);
  _Float16* W4p    = (_Float16*)(ws + OFF_W4);

  // Zero accumulators (summed + count are contiguous floats)
  {
    long nz = (long)NN * HID + NN;
    int blocks = (int)((nz + 255) / 256);
    zero_f32_kernel<<<blocks, 256, 0, stream>>>(summed, nz);
  }

  // Pack weights to f16 fragment order
  {
    int t1 = (IN1 / 32) * (HID / 16) * 512;
    pack_weight_kernel<<<(t1 + 255) / 256, 256, 0, stream>>>(W1, W1p, HID, HID / 16, t1);
    int t2 = (HID / 32) * (HID / 16) * 512;
    pack_weight_kernel<<<(t2 + 255) / 256, 256, 0, stream>>>(W2, W2p, HID, HID / 16, t2);
    int t3 = (IN2 / 32) * (HID / 16) * 512;
    pack_weight_kernel<<<(t3 + 255) / 256, 256, 0, stream>>>(W3, W3p, HID, HID / 16, t3);
    int t4 = (HID / 32) * (TGT / 16) * 512;
    pack_weight_kernel<<<(t4 + 255) / 256, 256, 0, stream>>>(W4, W4p, TGT, TGT / 16, t4);
  }

  // Edge MLP + scatter-add (800000 / 64 = 12500 blocks)
  edge_mlp_scatter_kernel<<<NE / EM, 256, 0, stream>>>(
      x, ei, ea, W1p, b1, W2p, b2, summed, count);

  // Node MLP (ceil(50000/32) = 1563 blocks)
  node_mlp_kernel<<<(NN + 31) / 32, 256, 0, stream>>>(
      x, u, batch, summed, count, W3p, b3, W4p, b4, out);
}
